// PairMixerBlock_79130477461837
// MI455X (gfx1250) — matrix operand order repaired and checked by reading, not compiled, not run on hardware
//
#include <hip/hip_runtime.h>
#include <hip/hip_bf16.h>

#define L   512
#define CZ  128
#define CH  128
#define NPIX (L*L)

typedef __attribute__((ext_vector_type(16))) __bf16 v16bf;
typedef __attribute__((ext_vector_type(8)))  __bf16 v8bf;
typedef __attribute__((ext_vector_type(8)))  float  v8f;
typedef __attribute__((ext_vector_type(4)))  float  v4f;

#define GAS __attribute__((address_space(1)))

__device__ __forceinline__ v8bf  gload8 (const __bf16* p){ return *(const GAS v8bf*)p; }
__device__ __forceinline__ void  gstore8(__bf16* p, v8bf v){ *(GAS v8bf*)p = v; }
__device__ __forceinline__ float gloadf (const float* p){ return *(const GAS float*)p; }
__device__ __forceinline__ void  gstoref(float* p, float v){ *(GAS float*)p = v; }
__device__ __forceinline__ v4f   gload4f(const float* p){ return *(const GAS v4f*)p; }
__device__ __forceinline__ __bf16 gloadbf(const __bf16* p){ return *(const GAS __bf16*)p; }
__device__ __forceinline__ void  gstorebf(__bf16* p, __bf16 v){ *(GAS __bf16*)p = v; }

__device__ __forceinline__ float sigmoidf_(float x){ return 1.f/(1.f+__expf(-x)); }

__device__ __forceinline__ v8f wmma_bf16(v16bf a, v16bf b, v8f c){
  return __builtin_amdgcn_wmma_f32_16x16x32_bf16(false, a, false, b, (short)0, c, false, false);
}

__device__ __forceinline__ v16bf cat8(v8bf lo, v8bf hi){
  return __builtin_shufflevector(lo, hi, 0,1,2,3,4,5,6,7,8,9,10,11,12,13,14,15);
}

// ---- fragment loaders -------------------------------------------------------
// A fragment: 16x32 (MxK) bf16 tile, row-major, ld multiple of 8.
// k order used here: lane<16 -> row=lane, K=0..15 ; lane>=16 -> row=lane-16, K=16..31
__device__ __forceinline__ v16bf frag_a_lds(const __bf16* src, int ld){
  int lane = threadIdx.x & 31;
  const __bf16* p = src + (size_t)(lane & 15)*ld + ((lane < 16) ? 0 : 16);
  return cat8(*(const v8bf*)p, *(const v8bf*)(p + 8));
}
__device__ __forceinline__ v16bf frag_a_g(const __bf16* src, int ld){
  int lane = threadIdx.x & 31;
  const __bf16* p = src + (size_t)(lane & 15)*ld + ((lane < 16) ? 0 : 16);
  return cat8(gload8(p), gload8(p + 8));
}
// B fragment from B^T row-major: src[n][k].
// k order used here: lane<16 -> col=lane, K=0..15 ; lane>=16 -> col=lane-16, K=16..31
__device__ __forceinline__ v16bf frag_b_nt_g(const __bf16* src, int ld){
  int lane = threadIdx.x & 31;
  const __bf16* p = src + (size_t)(lane & 15)*ld + ((lane < 16) ? 0 : 16);
  return cat8(gload8(p), gload8(p + 8));
}
// B fragment from pre-swizzled weight: tile = cb*nks + ks, 1024B/tile, contiguous/lane.
__device__ __forceinline__ v16bf frag_swz_g(const __bf16* swz, int tile){
  int lane = threadIdx.x & 31;
  const __bf16* p = swz + ((size_t)tile << 9) + lane*16;
  return cat8(gload8(p), gload8(p + 8));
}

// ---------------------------------------------------------------------------
// Prep: swizzle f32 KxN weight into bf16 fragment layout.
// ---------------------------------------------------------------------------
__global__ __launch_bounds__(256)
void k_swz(const float* __restrict__ w, __bf16* __restrict__ out, int K, int N)
{
  int o = blockIdx.x*256 + threadIdx.x;
  int e    = o & 15;
  int lane = (o >> 4) & 31;
  int t    = o >> 9;
  int nks  = K >> 5;
  int ks   = t % nks;
  int cb   = t / nks;
  int col  = cb*16 + (lane & 15);
  int k    = ks*32 + ((lane < 16) ? 0 : 16) + e;
  gstorebf(out + o, (__bf16)gloadf(w + (size_t)k*N + col));
}

// ---------------------------------------------------------------------------
// Per-channel 512x512 bf16 transpose (incoming tri-mul operands)
// ---------------------------------------------------------------------------
__global__ __launch_bounds__(256)
void k_tr2d(const __bf16* __restrict__ src, __bf16* __restrict__ dst)
{
  __shared__ __attribute__((aligned(16))) __bf16 tile[64][72];
  const int c   = blockIdx.z;
  const int rb0 = blockIdx.y * 64;
  const int cb0 = blockIdx.x * 64;
  const int tid = threadIdx.x;
  const size_t plane = (size_t)c * L * L;
  #pragma unroll
  for(int v=0; v<2; ++v){
    int lin = tid + v*256;
    int r   = lin >> 3;
    int c8  = lin & 7;
    v8bf d = gload8(src + plane + (size_t)(rb0 + r)*L + cb0 + c8*8);
    #pragma unroll
    for(int e=0;e<8;++e) tile[c8*8+e][r] = d[e];
  }
  __syncthreads();
  #pragma unroll
  for(int v=0; v<2; ++v){
    int lin = tid + v*256;
    int xr  = lin >> 3;
    int y8  = lin & 7;
    v8bf d;
    #pragma unroll
    for(int e=0;e<8;++e) d[e] = tile[xr][y8*8+e];
    gstore8(dst + plane + (size_t)(cb0 + xr)*L + rb0 + y8*8, d);
  }
}

// ---------------------------------------------------------------------------
// K1 helper: one gated projection pair (p * sigmoid(g) * mask), 16B store/lane
// ---------------------------------------------------------------------------
__device__ __forceinline__ void proj_pair(
    const v16bf afr[4], const __bf16* swzP, const float* bP,
    const __bf16* swzG, const float* bG, const float* mask,
    __bf16* dst, int pixbase, int rb, int cb0, int n, int Moff, int p0, int q0)
{
  for(int t=0;t<4;++t){
    int cb = cb0 + t;
    v8f accp = {}; v8f accq = {};
    #pragma unroll
    for(int ks=0;ks<4;++ks){
      accp = wmma_bf16(afr[ks], frag_swz_g(swzP, cb*4+ks), accp);
      accq = wmma_bf16(afr[ks], frag_swz_g(swzG, cb*4+ks), accq);
    }
    float biasp = gloadf(bP + cb*16 + n);
    float biasq = gloadf(bG + cb*16 + n);
    int c = cb*16 + n;
    v8bf pk;
    #pragma unroll
    for(int v=0; v<8; ++v){
      float mv = gloadf(mask + pixbase + rb*16 + Moff + v);
      pk[v] = (__bf16)((accp[v] + biasp) * sigmoidf_(accq[v] + biasq) * mv);
    }
    gstore8(dst + ((size_t)c*L + p0)*L + q0, pk);
  }
}

// ---------------------------------------------------------------------------
// K1: LN + a=(ap)*sig(ag)*mask, b=(bp)*sig(bg)*mask, g=sig(gp)
//     a,b stored channel-major [c][p][q]
// ---------------------------------------------------------------------------
__global__ __launch_bounds__(256)
void k_proj(const float* __restrict__ z, const float* __restrict__ mask,
            const float* ln_g, const float* ln_b,
            const __bf16* sap, const float* b_ap,
            const __bf16* sag, const float* b_ag,
            const __bf16* sbp, const float* b_bp,
            const __bf16* sbg, const float* b_bg,
            const __bf16* sgg, const float* b_gg,
            __bf16* __restrict__ a_dst, __bf16* __restrict__ b_dst,
            __bf16* __restrict__ g_out)
{
  __shared__ float  zst[64][129];
  __shared__ __attribute__((aligned(16))) __bf16 zl[64][136];
  const int tid = threadIdx.x;
  const int pixbase = blockIdx.x * 64;

  for(int i = tid; i < 64*CZ/4; i += 256){
    v4f d = gload4f(z + (size_t)pixbase*CZ + i*4);
    int r = (i*4) >> 7, cc = (i*4) & 127;
    #pragma unroll
    for(int e=0;e<4;++e) zst[r][cc+e] = d[e];
  }
  __syncthreads();

  if(tid < 64){
    float m = 0.f;
    for(int c=0;c<CZ;++c) m += zst[tid][c];
    m *= (1.f/CZ);
    float v = 0.f;
    for(int c=0;c<CZ;++c){ float d = zst[tid][c]-m; v += d*d; }
    v *= (1.f/CZ);
    float inv = __frsqrt_rn(v + 1e-5f);
    for(int c=0;c<CZ;++c)
      zl[tid][c] = (__bf16)((zst[tid][c]-m)*inv*gloadf(ln_g+c) + gloadf(ln_b+c));
  }
  __syncthreads();

  const int wave = tid >> 5, lane = tid & 31;
  const int rb   = wave & 3;
  const int cb0  = (wave >> 2) * 4;
  const int n    = lane & 15;
  const int Moff = (lane < 16) ? 0 : 8;
  const int p0   = pixbase >> 9;
  const int q0   = (pixbase & 511) + rb*16 + Moff;

  v16bf afr[4];
  #pragma unroll
  for(int ks=0;ks<4;++ks) afr[ks] = frag_a_lds(&zl[rb*16][ks*32], 136);

  proj_pair(afr, sap, b_ap, sag, b_ag, mask, a_dst, pixbase, rb, cb0, n, Moff, p0, q0);
  proj_pair(afr, sbp, b_bp, sbg, b_bg, mask, b_dst, pixbase, rb, cb0, n, Moff, p0, q0);

  // gate g = sigmoid(zl @ w_g + b_g), stored [pix][c] (coalesced across lanes)
  for(int t=0;t<4;++t){
    int cb = cb0 + t;
    v8f acc = {};
    #pragma unroll
    for(int ks=0;ks<4;++ks)
      acc = wmma_bf16(afr[ks], frag_swz_g(sgg, cb*4+ks), acc);
    float bias = gloadf(b_gg + cb*16 + n);
    #pragma unroll
    for(int v=0; v<8; ++v){
      int pix = pixbase + rb*16 + Moff + v;
      gstorebf(g_out + (size_t)pix*CZ + cb*16 + n, (__bf16)sigmoidf_(acc[v] + bias));
    }
  }
}

// ---------------------------------------------------------------------------
// K2: batched NT GEMM per channel: x[i][c][j] = sum_k a_t[c][i][k]*b_t[c][j][k]
// 128x128 block tile; wave = 2 rowbands x 4 colbands (8 WMMA per 6 frag loads)
// ---------------------------------------------------------------------------
__global__ __launch_bounds__(256)
void k_tri(const __bf16* __restrict__ a_t, const __bf16* __restrict__ b_t,
           float* __restrict__ x)
{
  const int c     = blockIdx.z;
  const int ibase = blockIdx.y * 128;
  const int jbase = blockIdx.x * 128;
  const int wave  = threadIdx.x >> 5, lane = threadIdx.x & 31;
  const int rb0   = (wave & 3) * 2;     // rowbands rb0, rb0+1
  const int cb0   = (wave >> 2) * 4;    // colbands cb0..cb0+3
  const int n     = lane & 15;
  const int Moff  = (lane < 16) ? 0 : 8;

  const __bf16* A0 = a_t + ((size_t)c*L + ibase + (rb0+0)*16)*L;
  const __bf16* A1 = a_t + ((size_t)c*L + ibase + (rb0+1)*16)*L;
  const __bf16* B0 = b_t + ((size_t)c*L + jbase + cb0*16)*L;

  v8f acc[2][4] = {{{},{},{},{}},{{},{},{},{}}};
  for(int ks=0; ks<16; ++ks){
    v16bf a0 = frag_a_g(A0 + ks*32, L);
    v16bf a1 = frag_a_g(A1 + ks*32, L);
    if(ks < 15){
      __builtin_prefetch(A0 + (ks+1)*32, 0, 3);
      __builtin_prefetch(A1 + (ks+1)*32, 0, 3);
    }
    #pragma unroll
    for(int t=0;t<4;++t){
      v16bf bf = frag_b_nt_g(B0 + (size_t)t*16*L + ks*32, L);
      acc[0][t] = wmma_bf16(a0, bf, acc[0][t]);
      acc[1][t] = wmma_bf16(a1, bf, acc[1][t]);
    }
  }
  #pragma unroll
  for(int r=0;r<2;++r){
    #pragma unroll
    for(int t=0;t<4;++t){
      #pragma unroll
      for(int v=0;v<8;++v){
        int i = ibase + (rb0+r)*16 + Moff + v;
        int j = jbase + (cb0+t)*16 + n;       // lanes 0..15 -> contiguous j
        gstoref(x + ((size_t)i*CH + c)*L + j, acc[r][t][v]);
      }
    }
  }
}

// ---------------------------------------------------------------------------
// K3: LN(x) @ w_z + b_z, times gate, residual add into z.  x layout [i][c][j].
// ---------------------------------------------------------------------------
__global__ __launch_bounds__(256)
void k_out(const float* __restrict__ x, const __bf16* __restrict__ g_buf,
           const float* ln_g, const float* ln_b,
           const __bf16* swz, const float* b_z,
           float* __restrict__ zcur)
{
  __shared__ float  xst[64][129];
  __shared__ __attribute__((aligned(16))) __bf16 xl[64][136];
  const int tid = threadIdx.x;
  const int pixbase = blockIdx.x * 64;
  const int i0 = pixbase >> 9, j0 = pixbase & 511;
  for(int idx = tid; idx < CH*16; idx += 256){
    int c = idx >> 4, q4 = idx & 15;
    v4f d = gload4f(x + ((size_t)i0*CH + c)*L + j0 + q4*4);
    #pragma unroll
    for(int e=0;e<4;++e) xst[q4*4+e][c] = d[e];
  }
  __syncthreads();
  if(tid < 64){
    float m=0.f; for(int c=0;c<CH;++c) m += xst[tid][c]; m *= (1.f/CH);
    float v=0.f; for(int c=0;c<CH;++c){ float d=xst[tid][c]-m; v += d*d; } v *= (1.f/CH);
    float inv = __frsqrt_rn(v + 1e-5f);
    for(int c=0;c<CH;++c)
      xl[tid][c] = (__bf16)((xst[tid][c]-m)*inv*gloadf(ln_g+c) + gloadf(ln_b+c));
  }
  __syncthreads();
  const int wave=tid>>5, lane=tid&31;
  const int rb = wave & 3, cb0 = (wave>>2)*4;
  const int n = lane & 15, Moff = (lane<16)?0:8;
  v16bf afr[4];
  #pragma unroll
  for(int ks=0;ks<4;++ks) afr[ks] = frag_a_lds(&xl[rb*16][ks*32], 136);
  for(int t=0;t<4;++t){
    int cb = cb0 + t;
    v8f acc = {};
    #pragma unroll
    for(int ks=0;ks<4;++ks)
      acc = wmma_bf16(afr[ks], frag_swz_g(swz, cb*4+ks), acc);
    float bias = gloadf(b_z + cb*16 + n);
    #pragma unroll
    for(int v=0;v<8;++v){
      int pix = pixbase + rb*16 + Moff + v;
      size_t idx = (size_t)pix*CZ + cb*16 + n;   // lanes 0..15 -> contiguous c
      float zv = gloadf(zcur + idx);
      float gv = (float)gloadbf(g_buf + idx);
      gstoref(zcur + idx, zv + (acc[v] + bias)*gv);
    }
  }
}

// ---------------------------------------------------------------------------
// K4: transition: z += relu(LN(z)@w1+b1) @ w2 + b2
// ---------------------------------------------------------------------------
__global__ __launch_bounds__(256)
void k_trans(const float* ln_g, const float* ln_b,
             const __bf16* sw1, const float* b1,
             const __bf16* sw2, const float* b2,
             float* __restrict__ zcur)
{
  __shared__ float  zst [32][129];
  __shared__ __attribute__((aligned(16))) __bf16 zl  [32][136];
  __shared__ __attribute__((aligned(16))) __bf16 hbuf[32][520];
  const int tid = threadIdx.x;
  const int pixbase = blockIdx.x * 32;
  for(int i=tid;i<32*CZ/4;i+=256){
    v4f d = gload4f(zcur + (size_t)pixbase*CZ + i*4);
    int r = (i*4) >> 7, cc = (i*4) & 127;
    #pragma unroll
    for(int e=0;e<4;++e) zst[r][cc+e] = d[e];
  }
  __syncthreads();
  if(tid < 32){
    float m=0.f; for(int c=0;c<CZ;++c) m += zst[tid][c]; m *= (1.f/CZ);
    float v=0.f; for(int c=0;c<CZ;++c){ float d=zst[tid][c]-m; v += d*d; } v *= (1.f/CZ);
    float inv = __frsqrt_rn(v + 1e-5f);
    for(int c=0;c<CZ;++c)
      zl[tid][c] = (__bf16)((zst[tid][c]-m)*inv*gloadf(ln_g+c) + gloadf(ln_b+c));
  }
  __syncthreads();
  const int wave=tid>>5, lane=tid&31;
  const int n = lane & 15, Moff = (lane<16)?0:8;
  const int rb  = wave & 1;
  const int grp = wave >> 1;
  // h = relu(zl @ w1 + b1) : [32 x 512]   (w1 swizzled: nks=4, 32 col bands)
  {
    v16bf afr[4];
    #pragma unroll
    for(int ks=0;ks<4;++ks) afr[ks] = frag_a_lds(&zl[rb*16][ks*32], 136);
    for(int t=0;t<8;++t){
      int cb = grp*8 + t;
      v8f acc = {};
      #pragma unroll
      for(int ks=0;ks<4;++ks)
        acc = wmma_bf16(afr[ks], frag_swz_g(sw1, cb*4+ks), acc);
      float bias = gloadf(b1 + cb*16 + n);
      #pragma unroll
      for(int v=0;v<8;++v){
        float h = acc[v] + bias;
        hbuf[rb*16 + Moff + v][cb*16 + n] = (__bf16)(h > 0.f ? h : 0.f);
      }
    }
  }
  __syncthreads();
  // out = h @ w2 + b2 ; z += out        (w2 swizzled: nks=16, 8 col bands)
  {
    for(int t=0;t<2;++t){
      int cb = grp*2 + t;
      v8f acc = {};
      for(int ks=0;ks<16;++ks){
        v16bf af = frag_a_lds(&hbuf[rb*16][ks*32], 520);
        acc = wmma_bf16(af, frag_swz_g(sw2, cb*16+ks), acc);
      }
      float bias = gloadf(b2 + cb*16 + n);
      #pragma unroll
      for(int v=0;v<8;++v){
        int row = rb*16 + Moff + v;
        int c   = cb*16 + n;
        gstoref(zcur + (size_t)(pixbase+row)*CZ + c, zst[row][c] + acc[v] + bias);
      }
    }
  }
}

// ---------------------------------------------------------------------------
// param index map (recursive insertion order):
// 0:z 1:mask | tmi@2, tmo@18: [ln_in_g,ln_in_b,w_ap,b_ap,w_ag,b_ag,w_bp,b_bp,
//  w_bg,b_bg,ln_out_g,ln_out_b,w_z,b_z,w_g,b_g] | tr@34: [ln_g,ln_b,w1,b1,w2,b2]
// ---------------------------------------------------------------------------
extern "C" void kernel_launch(void* const* d_in, const int* in_sizes, int n_in,
                              void* d_out, int out_size, void* d_ws, size_t ws_size,
                              hipStream_t stream)
{
  const float* z0   = (const float*)d_in[0];
  const float* mask = (const float*)d_in[1];
  float* zcur = (float*)d_out;

  char* ws = (char*)d_ws;
  __bf16* a_t = (__bf16*)ws; ws += (size_t)CH*L*L*sizeof(__bf16);
  __bf16* b_t = (__bf16*)ws; ws += (size_t)CH*L*L*sizeof(__bf16);
  __bf16* g_b = (__bf16*)ws; ws += (size_t)NPIX*CZ*sizeof(__bf16);
  float*  x   = (float*) ws; ws += (size_t)NPIX*CH*sizeof(float);
  // raw (pre-transpose) staging reuses x's space during the incoming pass
  __bf16* a_raw = (__bf16*)x;
  __bf16* b_raw = a_raw + (size_t)CH*L*L;

  // swizzled weights
  __bf16* swz = (__bf16*)ws;
  __bf16* S[2][6];
  {
    size_t off = 0;
    for(int p=0;p<2;++p)
      for(int m=0;m<6;++m){ S[p][m] = swz + off; off += 16384; }
  }
  __bf16* SW1 = swz + 2*6*16384;
  __bf16* SW2 = SW1 + 65536;

  // ---- weight prep ----
  for(int pass=0; pass<2; ++pass){
    const float* const* P = (const float* const*)(d_in + 2 + pass*16);
    k_swz<<<64, 256, 0, stream>>>(P[2],  S[pass][0], 128, 128);  // w_ap
    k_swz<<<64, 256, 0, stream>>>(P[4],  S[pass][1], 128, 128);  // w_ag
    k_swz<<<64, 256, 0, stream>>>(P[6],  S[pass][2], 128, 128);  // w_bp
    k_swz<<<64, 256, 0, stream>>>(P[8],  S[pass][3], 128, 128);  // w_bg
    k_swz<<<64, 256, 0, stream>>>(P[14], S[pass][4], 128, 128);  // w_g
    k_swz<<<64, 256, 0, stream>>>(P[12], S[pass][5], 128, 128);  // w_z
  }
  const float* const* T = (const float* const*)(d_in + 34);
  k_swz<<<256, 256, 0, stream>>>(T[2], SW1, 128, 512);           // w1
  k_swz<<<256, 256, 0, stream>>>(T[4], SW2, 512, 128);           // w2

  hipMemcpyAsync(zcur, z0, (size_t)NPIX*CZ*sizeof(float),
                 hipMemcpyDeviceToDevice, stream);

  for(int pass=0; pass<2; ++pass){
    const float* const* P = (const float* const*)(d_in + 2 + pass*16);
    const int incoming = (pass == 0);          // tmi first, then tmo
    __bf16* adst = incoming ? a_raw : a_t;
    __bf16* bdst = incoming ? b_raw : b_t;
    k_proj<<<NPIX/64, 256, 0, stream>>>(zcur, mask,
        P[0], P[1],
        S[pass][0], P[3], S[pass][1], P[5],
        S[pass][2], P[7], S[pass][3], P[9],
        S[pass][4], P[15],
        adst, bdst, g_b);
    if(incoming){
      k_tr2d<<<dim3(8,8,CH), 256, 0, stream>>>(a_raw, a_t);
      k_tr2d<<<dim3(8,8,CH), 256, 0, stream>>>(b_raw, b_t);
    }
    k_tri<<<dim3(4, 4, CH), 256, 0, stream>>>(a_t, b_t, x);
    k_out<<<NPIX/64, 256, 0, stream>>>(x, g_b, P[10], P[11],
                                       S[pass][5], P[13], zcur);
  }
  k_trans<<<NPIX/32, 256, 0, stream>>>(T[0], T[1], SW1, T[3], SW2, T[5], zcur);
}
